// flash_attention_9878424781188
// MI455X (gfx1250) — matrix operand order repaired and checked by reading, not compiled, not run on hardware
//
#include <hip/hip_runtime.h>

// Flash attention fwd: softmax(Q K^T / sqrt(D)) V, N=8192, D=128, fp32 I/O.
// MI455X / gfx1250 (wave32) strategy:
//  * compute-bound (34 GFLOP, 17MB working set, fully L2-resident @192MB)
//  * fp32 -> bf16 hi+lo split, 3-term WMMA (hi*hi + hi*lo + lo*hi) chained
//    into one f32 accumulator => near-fp32 accuracy at bf16 WMMA rate
//  * one-time preprocessing writes K/V bf16 hi/lo planes (V transposed,
//    per-key-block tiled) into d_ws when ws_size allows -> staging is pure
//    copies (GLOBAL_LOAD_ASYNC_TO_LDS_B128 when available, ASYNCcnt-tracked)
//  * double-buffered LDS K/V tiles: one barrier per iteration, staging of
//    block kb+1 overlaps WMMA compute of block kb
//  * online softmax in exp2 domain (scale*log2e folded into Q),
//    row reductions via shfl_xor(1,2,4,8) over the 16-lane wave32 halves

#define FA_N      8192
#define FA_D      128
#define KBLK      32
#define NWAVES    4
#define NTHREADS  (NWAVES * 32)
#define NBLOCKS_K (FA_N / KBLK)

typedef __attribute__((ext_vector_type(16))) __bf16 v16bf;
typedef __attribute__((ext_vector_type(8)))  float  v8f;

union Frag {
  v16bf v;
  unsigned int u[8];
};

__device__ __forceinline__ v8f wmma_bf16(Frag a, Frag b, v8f c) {
  return __builtin_amdgcn_wmma_f32_16x16x32_bf16(false, a.v, false, b.v,
                                                 (short)0, c, false, false);
}

// Split two fp32 into packed bf16 hi-pair / lo-pair using v_perm_b32 packing.
__device__ __forceinline__ void split2(float f0, float f1,
                                       unsigned int &hi, unsigned int &lo) {
  unsigned int b0 = __float_as_uint(f0), b1 = __float_as_uint(f1);
  float r0 = f0 - __uint_as_float(b0 & 0xFFFF0000u);
  float r1 = f1 - __uint_as_float(b1 & 0xFFFF0000u);
  hi = __builtin_amdgcn_perm(b1, b0, 0x07060302u);
  lo = __builtin_amdgcn_perm(__float_as_uint(r1), __float_as_uint(r0), 0x07060302u);
}

#if __has_builtin(__builtin_amdgcn_global_load_async_to_lds_b128)
#define HAVE_ASYNC_LDS 1
__device__ __forceinline__ void async_copy16(const void* g, void* l) {
  // Builtin expects int4-vector pointers: global (AS1) src, LDS (AS3) dst.
  typedef int v4i_ __attribute__((vector_size(16)));
  typedef __attribute__((address_space(1))) v4i_* gp_t;
  typedef __attribute__((address_space(3))) v4i_* lp_t;
  __builtin_amdgcn_global_load_async_to_lds_b128((gp_t)g, (lp_t)l, 0, 0);
}
__device__ __forceinline__ void wait_async() {
  asm volatile("s_wait_asynccnt 0x0" ::: "memory");
}
#else
#define HAVE_ASYNC_LDS 0
__device__ __forceinline__ void wait_async() {}
#endif

// ---------------- preprocessing: K -> bf16 hi/lo planes (row-major) ----------
__global__ __launch_bounds__(256, 1)
void prep_k_kernel(const float* __restrict__ Km,
                   unsigned short* __restrict__ wsKhi,
                   unsigned short* __restrict__ wsKlo) {
  const size_t idx = (size_t)blockIdx.x * 256 + threadIdx.x;  // float4 index
  float4 f = *(const float4*)(Km + idx * 4);
  unsigned int h01, l01, h23, l23;
  split2(f.x, f.y, h01, l01);
  split2(f.z, f.w, h23, l23);
  *(uint2*)(wsKhi + idx * 4) = make_uint2(h01, h23);
  *(uint2*)(wsKlo + idx * 4) = make_uint2(l01, l23);
}

// ------- preprocessing: V -> transposed bf16 hi/lo, tiled per key-block -----
// Output layout: ws[((kb*128 + d) * 32) + k_local]  (kb = n/32)
__global__ __launch_bounds__(256, 1)
void prep_v_kernel(const float* __restrict__ Vm,
                   unsigned short* __restrict__ wsVthi,
                   unsigned short* __restrict__ wsVtlo) {
  __shared__ unsigned short tH[32][33];
  __shared__ unsigned short tL[32][33];
  const int tx = threadIdx.x;          // 0..31
  const int ty = threadIdx.y;          // 0..7
  const int n0 = blockIdx.x * 32;      // key tile
  const int d0 = blockIdx.y * 32;      // feature tile
#pragma unroll
  for (int r = 0; r < 4; ++r) {
    const int ny = ty + 8 * r;
    float f = Vm[(size_t)(n0 + ny) * FA_D + d0 + tx];
    unsigned int b = __float_as_uint(f);
    unsigned int h = b & 0xFFFF0000u;
    float res = f - __uint_as_float(h);
    tH[ny][tx] = (unsigned short)(h >> 16);
    tL[ny][tx] = (unsigned short)(__float_as_uint(res) >> 16);
  }
  __syncthreads();
#pragma unroll
  for (int r = 0; r < 4; ++r) {
    const int dy = ty + 8 * r;
    const size_t o = ((size_t)(n0 >> 5) * FA_D + d0 + dy) * 32 + tx;
    wsVthi[o] = tH[tx][dy];
    wsVtlo[o] = tL[tx][dy];
  }
}

// ------------------------------ main kernel ---------------------------------
template <bool USE_WS>
__global__ __launch_bounds__(NTHREADS, 1)
void fa_wmma_bf16split_kernel(const float* __restrict__ Q,
                              const float* __restrict__ Km,
                              const float* __restrict__ Vm,
                              float* __restrict__ Om,
                              const unsigned short* __restrict__ wsKhi,
                              const unsigned short* __restrict__ wsKlo,
                              const unsigned short* __restrict__ wsVthi,
                              const unsigned short* __restrict__ wsVtlo)
{
  // Double-buffered K (row-major [k][d]) and V (transposed [d][k]) tiles.
  __shared__ unsigned short sKhi[2][KBLK][FA_D + 8];   // row = 272B (16B-aligned)
  __shared__ unsigned short sKlo[2][KBLK][FA_D + 8];
  __shared__ unsigned short sVhi[2][FA_D][KBLK + 8];   // row = 80B (16B-aligned)
  __shared__ unsigned short sVlo[2][FA_D][KBLK + 8];
  __shared__ unsigned short sPhi[NWAVES][16][KBLK + 8];
  __shared__ unsigned short sPlo[NWAVES][16][KBLK + 8];

  const int tid  = threadIdx.x;
  const int wave = tid >> 5;
  const int lane = tid & 31;
  const int half = lane >> 4;   // which 16-lane half of the wave
  const int l16  = lane & 15;

  const int qrow0 = (blockIdx.x * NWAVES + wave) * 16;

  // 1/sqrt(128) * log2(e): softmax done in exp2 domain
  const float qscale = 0.08838834764831845f * 1.4426950408889634f;

  // Stage key-block kbs into LDS buffer b.
  auto stage = [&](int kbs, int b) {
    if (USE_WS) {
#if HAVE_ASYNC_LDS
      const char* kh = (const char*)(wsKhi + (size_t)kbs * KBLK * FA_D);
      const char* kl = (const char*)(wsKlo + (size_t)kbs * KBLK * FA_D);
      const char* vh = (const char*)(wsVthi + (size_t)kbs * FA_D * KBLK);
      const char* vl = (const char*)(wsVtlo + (size_t)kbs * FA_D * KBLK);
#pragma unroll
      for (int i = 0; i < 4; ++i) {
        int c = tid + i * NTHREADS;
        int kr = c >> 4, ko = (c & 15) * 8;   // K: 16 chunks per row
        int d  = c >> 2, vo = (c & 3) * 8;    // V: 4 chunks per d-row
        async_copy16(kh + c * 16, &sKhi[b][kr][ko]);
        async_copy16(kl + c * 16, &sKlo[b][kr][ko]);
        async_copy16(vh + c * 16, &sVhi[b][d][vo]);
        async_copy16(vl + c * 16, &sVlo[b][d][vo]);
      }
#else
      const uint4* kh = (const uint4*)(wsKhi + (size_t)kbs * KBLK * FA_D);
      const uint4* kl = (const uint4*)(wsKlo + (size_t)kbs * KBLK * FA_D);
      const uint4* vh = (const uint4*)(wsVthi + (size_t)kbs * FA_D * KBLK);
      const uint4* vl = (const uint4*)(wsVtlo + (size_t)kbs * FA_D * KBLK);
      uint4 t[16];
#pragma unroll
      for (int i = 0; i < 4; ++i) {
        int c = tid + i * NTHREADS;
        t[i]      = kh[c];
        t[4 + i]  = kl[c];
        t[8 + i]  = vh[c];
        t[12 + i] = vl[c];
      }
#pragma unroll
      for (int i = 0; i < 4; ++i) {
        int c = tid + i * NTHREADS;
        int kr = c >> 4, ko = (c & 15) * 8;
        int d  = c >> 2, vo = (c & 3) * 8;
        *(uint4*)&sKhi[b][kr][ko] = t[i];
        *(uint4*)&sKlo[b][kr][ko] = t[4 + i];
        *(uint4*)&sVhi[b][d][vo]  = t[8 + i];
        *(uint4*)&sVlo[b][d][vo]  = t[12 + i];
      }
#endif
    } else {
      // fallback: stage + convert from fp32 in-kernel
      {
        const float* kp = Km + (size_t)kbs * KBLK * FA_D;
#pragma unroll
        for (int i = 0; i < (KBLK * FA_D) / (4 * NTHREADS); ++i) {
          int idx = tid + i * NTHREADS;      // float4 index
          int kr  = idx >> 5;                // 32 float4 per row
          int dq  = (idx & 31) * 4;
          float4 f = *(const float4*)(kp + kr * FA_D + dq);
          unsigned int h01, l01, h23, l23;
          split2(f.x, f.y, h01, l01);
          split2(f.z, f.w, h23, l23);
          *(uint2*)&sKhi[b][kr][dq] = make_uint2(h01, h23);
          *(uint2*)&sKlo[b][kr][dq] = make_uint2(l01, l23);
        }
      }
      {
        const float* vp = Vm + (size_t)kbs * KBLK * FA_D + tid;  // d = tid
#pragma unroll
        for (int q = 0; q < KBLK / 4; ++q) {
          float f0 = vp[(4 * q + 0) * FA_D];
          float f1 = vp[(4 * q + 1) * FA_D];
          float f2 = vp[(4 * q + 2) * FA_D];
          float f3 = vp[(4 * q + 3) * FA_D];
          unsigned int h01, l01, h23, l23;
          split2(f0, f1, h01, l01);
          split2(f2, f3, h23, l23);
          *(uint2*)&sVhi[b][tid][4 * q] = make_uint2(h01, h23);
          *(uint2*)&sVlo[b][tid][4 * q] = make_uint2(l01, l23);
        }
      }
    }
  };

  // ---- Q fragments (16-bit A layout), scaled + hi/lo split: 4 K=32 slices
  Frag qhi[4], qlo[4];
  {
    const float* qp = Q + (size_t)(qrow0 + l16) * FA_D;
#pragma unroll
    for (int s = 0; s < 4; ++s) {
#pragma unroll
      for (int j = 0; j < 8; ++j) {
        // contiguous k order (matches B frags): lane<16 holds K {0..15}, lane>=16 holds {16..31}
        const int kloc = half * 16 + 2 * j;
        const int d = s * 32 + kloc;
        split2(qp[d] * qscale, qp[d + 1] * qscale, qhi[s].u[j], qlo[s].u[j]);
      }
    }
  }

  // ---- accumulators: O (8 D-tiles of 16 cols), row max m, row sum l
  v8f o[8];
  float m[8], l[8];
#pragma unroll
  for (int t = 0; t < 8; ++t)
#pragma unroll
    for (int j = 0; j < 8; ++j) o[t][j] = 0.0f;
#pragma unroll
  for (int j = 0; j < 8; ++j) { m[j] = -3.0e38f; l[j] = 0.0f; }

  stage(0, 0);  // prologue: fill buffer 0

  for (int kb = 0; kb < NBLOCKS_K; ++kb) {
    const int cur = kb & 1;
    wait_async();     // our staged copies (issued last iteration) are done
    __syncthreads();  // everyone's staging visible; old buffer free to overwrite

    if (kb + 1 < NBLOCKS_K) stage(kb + 1, cur ^ 1);

    // ---- S = (Q*scale) K^T for 16x32 score tile: 2 accumulators x 12 WMMA
    v8f s0, s1;
#pragma unroll
    for (int j = 0; j < 8; ++j) { s0[j] = 0.0f; s1[j] = 0.0f; }

#pragma unroll
    for (int s = 0; s < 4; ++s) {
      Frag bh, bl;
      {  // keys kb*32 + 0..15 : B layout = contiguous 16 d-values per lane
        const unsigned int* ph = (const unsigned int*)&sKhi[cur][l16][s * 32 + half * 16];
        const unsigned int* pl = (const unsigned int*)&sKlo[cur][l16][s * 32 + half * 16];
#pragma unroll
        for (int j = 0; j < 8; ++j) { bh.u[j] = ph[j]; bl.u[j] = pl[j]; }
      }
      s0 = wmma_bf16(qhi[s], bh, s0);
      s0 = wmma_bf16(qhi[s], bl, s0);
      s0 = wmma_bf16(qlo[s], bh, s0);
      {  // keys kb*32 + 16..31
        const unsigned int* ph = (const unsigned int*)&sKhi[cur][16 + l16][s * 32 + half * 16];
        const unsigned int* pl = (const unsigned int*)&sKlo[cur][16 + l16][s * 32 + half * 16];
#pragma unroll
        for (int j = 0; j < 8; ++j) { bh.u[j] = ph[j]; bl.u[j] = pl[j]; }
      }
      s1 = wmma_bf16(qhi[s], bh, s1);
      s1 = wmma_bf16(qhi[s], bl, s1);
      s1 = wmma_bf16(qlo[s], bh, s1);
    }

    // ---- online softmax (exp2 domain); C layout: lane=col, VGPR j=row j+8*half
    float alpha[8];
#pragma unroll
    for (int j = 0; j < 8; ++j) {
      float mx = fmaxf(s0[j], s1[j]);
      mx = fmaxf(mx, __shfl_xor(mx, 1, 32));
      mx = fmaxf(mx, __shfl_xor(mx, 2, 32));
      mx = fmaxf(mx, __shfl_xor(mx, 4, 32));
      mx = fmaxf(mx, __shfl_xor(mx, 8, 32));
      float mn = fmaxf(m[j], mx);
      alpha[j] = exp2f(m[j] - mn);
      m[j] = mn;
      float p0 = exp2f(s0[j] - mn);
      float p1 = exp2f(s1[j] - mn);
      float rs = p0 + p1;
      rs += __shfl_xor(rs, 1, 32);
      rs += __shfl_xor(rs, 2, 32);
      rs += __shfl_xor(rs, 4, 32);
      rs += __shfl_xor(rs, 8, 32);
      l[j] = l[j] * alpha[j] + rs;

      // store P (hi/lo bf16) to wave-private LDS at [row][col]
      const int row = 8 * half + j;
      unsigned int b0 = __float_as_uint(p0) & 0xFFFF0000u;
      sPhi[wave][row][l16] = (unsigned short)(b0 >> 16);
      sPlo[wave][row][l16] =
          (unsigned short)(__float_as_uint(p0 - __uint_as_float(b0)) >> 16);
      unsigned int b1 = __float_as_uint(p1) & 0xFFFF0000u;
      sPhi[wave][row][16 + l16] = (unsigned short)(b1 >> 16);
      sPlo[wave][row][16 + l16] =
          (unsigned short)(__float_as_uint(p1 - __uint_as_float(b1)) >> 16);
    }

    // rescale O accumulators
#pragma unroll
    for (int t = 0; t < 8; ++t)
#pragma unroll
      for (int j = 0; j < 8; ++j) o[t][j] *= alpha[j];

    // wave-internal LDS RAW: keep compiler from reordering; hw LDS is in-order
    __builtin_amdgcn_fence(__ATOMIC_RELEASE, "workgroup");
    __builtin_amdgcn_wave_barrier();
    __builtin_amdgcn_fence(__ATOMIC_ACQUIRE, "workgroup");

    // ---- P fragments (A layout) from LDS
    Frag phi, plo;
#pragma unroll
    for (int j = 0; j < 8; ++j) {
      const int kloc = half * 16 + 2 * j;
      phi.u[j] = *(const unsigned int*)&sPhi[wave][l16][kloc];
      plo.u[j] = *(const unsigned int*)&sPlo[wave][l16][kloc];
    }

    // ---- O += P V : 8 D-tiles x 3 WMMA; V frag = contiguous 16 k per lane
#pragma unroll
    for (int t = 0; t < 8; ++t) {
      Frag vh, vl;
      const unsigned int* ph = (const unsigned int*)&sVhi[cur][t * 16 + l16][half * 16];
      const unsigned int* pl = (const unsigned int*)&sVlo[cur][t * 16 + l16][half * 16];
#pragma unroll
      for (int j = 0; j < 8; ++j) { vh.u[j] = ph[j]; vl.u[j] = pl[j]; }
      o[t] = wmma_bf16(phi, vh, o[t]);
      o[t] = wmma_bf16(phi, vl, o[t]);
      o[t] = wmma_bf16(plo, vh, o[t]);
    }
  }

  // ---- epilogue: O / l, direct fp32 stores (coalesced per 16-lane half)
#pragma unroll
  for (int j = 0; j < 8; ++j) {
    const float inv = 1.0f / l[j];
    const int row = qrow0 + 8 * half + j;
    float* op = Om + (size_t)row * FA_D + l16;
#pragma unroll
    for (int t = 0; t < 8; ++t) op[t * 16] = o[t][j] * inv;
  }
}

extern "C" void kernel_launch(void* const* d_in, const int* in_sizes, int n_in,
                              void* d_out, int out_size, void* d_ws, size_t ws_size,
                              hipStream_t stream) {
  (void)in_sizes; (void)n_in; (void)out_size;
  const float* Q = (const float*)d_in[0];
  const float* K = (const float*)d_in[1];
  const float* V = (const float*)d_in[2];
  float* O = (float*)d_out;

  const size_t planeElems = (size_t)FA_N * FA_D;          // u16 elements
  const size_t wsNeeded = 4 * planeElems * sizeof(unsigned short);  // 8 MB

  dim3 grid(FA_N / (16 * NWAVES));  // 128 blocks x 4 waves, 1 q-tile per wave

  if (ws_size >= wsNeeded) {
    unsigned short* wsKhi  = (unsigned short*)d_ws;
    unsigned short* wsKlo  = wsKhi + planeElems;
    unsigned short* wsVthi = wsKlo + planeElems;
    unsigned short* wsVtlo = wsVthi + planeElems;
    prep_k_kernel<<<dim3((FA_N * FA_D / 4) / 256), 256, 0, stream>>>(K, wsKhi, wsKlo);
    prep_v_kernel<<<dim3(FA_N / 32, FA_D / 32), dim3(32, 8), 0, stream>>>(V, wsVthi, wsVtlo);
    fa_wmma_bf16split_kernel<true><<<grid, NTHREADS, 0, stream>>>(
        Q, K, V, O, wsKhi, wsKlo, wsVthi, wsVtlo);
  } else {
    fa_wmma_bf16split_kernel<false><<<grid, NTHREADS, 0, stream>>>(
        Q, K, V, O, nullptr, nullptr, nullptr, nullptr);
  }
}
